// ABL_79783312490735
// MI455X (gfx1250) — compile-verified, matrix operand order checked by reading, not run on hardware
//
#include <hip/hip_runtime.h>
#include <math.h>

#define BB 4
#define HH 512
#define WW 512
#define CC 19
#define PIX (BB*HH*WW)
#define NBINS 128
#define NPART 1024

typedef __attribute__((ext_vector_type(16))) _Float16 v16h;
typedef __attribute__((ext_vector_type(8)))  float    v8f;

// ---------------------------------------------------------------- init
__global__ void init_kernel(int* hist, int* anyFlag) {
  int t = threadIdx.x;
  if (t < NBINS) hist[t] = 0;
  if (t < BB) anyFlag[t] = 0;
}

// ------------------------------------------------------ gt -> boundary
__global__ void gt_boundary_kernel(const int* __restrict__ tgt,
                                   unsigned char* __restrict__ gtb,
                                   int* __restrict__ anyFlag) {
  int idx = blockIdx.x * blockDim.x + threadIdx.x;
  if (idx >= PIX) return;
  int b = idx / (HH*WW);
  int r = idx % (HH*WW);
  int y = r / WW, x = r % WW;
  int g = tgt[idx];
  int dn = (y < HH-1) ? tgt[idx + WW] : g;
  int rt = (x < WW-1) ? tgt[idx + 1] : g;
  unsigned char bnd = (dn != g) || (rt != g) || (g == 255);
  gtb[idx] = bnd;
  if (bnd) anyFlag[b] = 1;   // idempotent racy store of 1: deterministic
}

// --------------------------------------------- EDT phase 1 (vertical)
__global__ void edt_vertical_kernel(const unsigned char* __restrict__ gtb,
                                    float* __restrict__ G) {
  int col = blockIdx.x * blockDim.x + threadIdx.x;
  if (col >= BB*WW) return;
  int b = col / WW, x = col % WW;
  const size_t base = (size_t)b * HH * WW + x;
  float carry = 1e6f;
  for (int y = 0; y < HH; ++y) {
    size_t i = base + (size_t)y * WW;
    float cv = gtb[i] ? 0.0f : 1e6f;
    carry = fminf(cv, carry + 1.0f);
    G[i] = carry;
  }
  carry = 1e6f;
  for (int y = HH-1; y >= 0; --y) {
    size_t i = base + (size_t)y * WW;
    float cv = gtb[i] ? 0.0f : 1e6f;
    carry = fminf(cv, carry + 1.0f);
    G[i] = fminf(G[i], carry);
  }
}

// ---------------------------------- EDT phase 2 (per-row min-plus, LDS)
__global__ __launch_bounds__(WW) void edt_row_kernel(
    const float* __restrict__ G, const int* __restrict__ anyFlag,
    float* __restrict__ dist) {
  __shared__ float g2[WW];
  int row = blockIdx.x;          // 0 .. B*H-1
  int b = row / HH;
  int j = threadIdx.x;
  size_t base = (size_t)row * WW;
  float g = G[base + j];
  g2[j] = g * g;
  __syncthreads();
  float best = 3.4e38f;
  #pragma unroll 4
  for (int k = 0; k < WW; ++k) {
    float diff = (float)(j - k);
    best = fminf(best, g2[k] + diff * diff);
  }
  float d = fmaxf(sqrtf(best) - 1.0f, 0.0f);
  if (!anyFlag[b]) d = 0.0f;
  dist[base + j] = d;
}

// ----------------------------------------------- KL(a,b) helper (C=19)
__device__ __forceinline__ float kl_pair(const float* __restrict__ bb,
                                         const float* __restrict__ aa,
                                         float mB, float sB) {
  // KL = sum softmax(b)*(log_softmax(b)-log_softmax(a))
  //    = sum p_b*(b-a) + lse(a) - lse(b)
  float mA = -3.4e38f;
  #pragma unroll
  for (int c = 0; c < CC; ++c) mA = fmaxf(mA, aa[c]);
  float sA = 0.0f;
  #pragma unroll
  for (int c = 0; c < CC; ++c) sA += __expf(aa[c] - mA);
  float lseA = mA + __logf(sA);
  float lseB = mB + __logf(sB);
  float dot = 0.0f, invS = 1.0f / sB;
  #pragma unroll
  for (int c = 0; c < CC; ++c) {
    float pb = __expf(bb[c] - mB) * invS;
    dot += pb * (bb[c] - aa[c]);
  }
  return dot + lseA - lseB;
}

// --------------------------- KL edge map + histogram (for eps search)
__global__ void kl_map_kernel(const float* __restrict__ logits,
                              float* __restrict__ klmap,
                              int* __restrict__ hist) {
  __shared__ int lhist[NBINS];
  int t = threadIdx.x;
  if (t < NBINS) lhist[t] = 0;
  __syncthreads();
  int idx = blockIdx.x * blockDim.x + t;
  if (idx < PIX) {
    int r = idx % (HH*WW);
    int y = r / WW, x = r % WW;
    const float* pc = logits + (size_t)idx * CC;
    __builtin_prefetch(pc + (size_t)WW * CC, 0, 1);  // down-neighbor row
    float mB = -3.4e38f;
    #pragma unroll
    for (int c = 0; c < CC; ++c) mB = fmaxf(mB, pc[c]);
    float sB = 0.0f;
    #pragma unroll
    for (int c = 0; c < CC; ++c) sB += __expf(pc[c] - mB);
    float kl = 0.0f;
    if (y < HH-1) kl += kl_pair(pc, pc + (size_t)WW*CC, mB, sB);  // ud
    if (x < WW-1) kl += kl_pair(pc, pc + CC,            mB, sB);  // lr
    klmap[idx] = kl;
    // bucket t: kl > eps_j (= 1e-5*1.2^j) iff j < t
    int bin;
    if (kl <= 1e-5f) bin = 0;
    else {
      int ti = (int)ceilf(__logf(kl * 1e5f) * 5.48486f); // 1/ln(1.2)
      bin = ti < 1 ? 1 : (ti > NBINS-1 ? NBINS-1 : ti);
    }
    atomicAdd(&lhist[bin], 1);
  }
  __syncthreads();
  if (t < NBINS) atomicAdd(&hist[t], lhist[t]);
}

// ----------------------- eps = 1e-5 * 1.2^k, smallest k with cnt<=maxN
__global__ void eps_kernel(const int* __restrict__ hist,
                           float* __restrict__ epsOut) {
  if (blockIdx.x == 0 && threadIdx.x == 0) {
    float maxN = (float)(HH * WW) * 0.01f;
    int cnt[NBINS + 1];
    cnt[NBINS] = 0;
    for (int tb = NBINS - 1; tb >= 0; --tb) cnt[tb] = cnt[tb+1] + hist[tb];
    int k = 0;
    while (k < NBINS - 1 && (float)cnt[k+1] > maxN) ++k;
    epsOut[0] = 1e-5f * __powf(1.2f, (float)k);
  }
}

// --------------------------------------- threshold + 3x3 dilation (OR)
__global__ void threshold_dilate_kernel(const float* __restrict__ klmap,
                                        const float* __restrict__ epsIn,
                                        unsigned char* __restrict__ pred) {
  int idx = blockIdx.x * blockDim.x + threadIdx.x;
  if (idx >= PIX) return;
  float eps = epsIn[0];
  int r = idx % (HH*WW);
  int y = r / WW, x = r % WW;
  int baseb = idx - r;   // b*H*W
  unsigned char v = 0;
  #pragma unroll
  for (int dy = -1; dy <= 1; ++dy)
    #pragma unroll
    for (int dx = -1; dx <= 1; ++dx) {
      int yy = y + dy, xx = x + dx;
      if (yy >= 0 && yy < HH && xx >= 0 && xx < WW)
        v |= (klmap[baseb + yy*WW + xx] > eps) ? 1 : 0;
    }
  pred[idx] = v;
}

// ------------------------------ direction loss: wave-per-pixel + WMMA
// 8-direction x 19-channel reductions done on the matrix core:
// A (16x32 f16) holds per-direction/per-channel terms, B = ones(32x16),
// D column n = row sums  =>  per-direction channel reduction.
// All masking is arithmetic (cndmask/mul), no per-element EXEC branches.
__global__ __launch_bounds__(256) void dir_loss_kernel(
    const float* __restrict__ logits,
    const float* __restrict__ dist,
    const unsigned char* __restrict__ pred,
    float* __restrict__ lossPart,
    int* __restrict__ cntPart) {
  __shared__ float shA[8][32];   // per-wave center logits (padded w/ zeros)
  __shared__ float shS[8][16];   // per-wave row-sum broadcast
  __shared__ float shLoss[8];
  __shared__ int   shCnt[8];

  const int lane = threadIdx.x & 31;
  const int wid  = threadIdx.x >> 5;
  const int nWaves = gridDim.x * 8;
  const int waveId = blockIdx.x * 8 + wid;

  const int row  = lane & 15;     // A-matrix row (direction)
  const int half = lane >> 4;     // K-chunk selector per ISA A layout
  const bool realRow = row < 8;
  const int d = realRow ? row : 0;

  const int dx9[9] = {1,-1,0,0,-1,1,-1,1,0};   // applied to H (rows)
  const int dy9[9] = {0,0,-1,1,1,1,-1,-1,0};   // applied to W (cols)

  // 16-bit A 16x32 layout: lanes 0-15 hold K 0-7,16-23; lanes 16-31 K 8-15,24-31
  int kmap[16];
  int kclamp[16];
  float kmask[16];
  #pragma unroll
  for (int e = 0; e < 16; ++e) {
    int k = e + ((e < 8) ? 0 : 8) + half * 8;
    kmap[e] = k;
    kclamp[e] = (k < CC) ? k : (CC - 1);
    kmask[e] = (realRow && k < CC) ? 1.0f : 0.0f;
  }

  v16h ones;
  #pragma unroll
  for (int e = 0; e < 16; ++e) ones[e] = (_Float16)1.0f;

  float waveLoss = 0.0f;
  int   waveCnt  = 0;

  for (int p = waveId; p < PIX; p += nWaves) {   // wave-uniform loop
    if (!pred[p]) continue;                      // uniform: EXEC stays full
    int r = p % (HH*WW);
    int b = p / (HH*WW);
    int y = r / WW, x = r % WW;

    // argmin over 9 padded distances (pad = MAX_DIS = 1e5, first-min wins)
    float dmin = 3.4e38f; int dirGt = 0;
    #pragma unroll
    for (int i = 0; i < 9; ++i) {
      int yy = y + dx9[i], xx = x + dy9[i];
      float dv = 1e5f;
      if (yy >= 0 && yy < HH && xx >= 0 && xx < WW)
        dv = dist[(size_t)b*HH*WW + (size_t)yy*WW + xx];
      if (dv < dmin) { dmin = dv; dirGt = i; }
    }
    if (dirGt == 8) continue;                    // uniform
    waveCnt++;
    float w = fminf(dist[p], 20.0f) * (1.0f / 20.0f);

    // center logits: lane = channel; wave-reduce lse(a)
    const float* pc = logits + (size_t)p * CC;
    float aRaw = pc[(lane < CC) ? lane : (CC - 1)];   // unconditional load
    float aV = (lane < CC) ? aRaw : -3.4e38f;
    shA[wid][lane] = (lane < CC) ? aRaw : 0.0f;
    float m = aV;
    #pragma unroll
    for (int o = 16; o > 0; o >>= 1) m = fmaxf(m, __shfl_xor(m, o, 32));
    float ev = (lane < CC) ? __expf(aV - m) : 0.0f;
    #pragma unroll
    for (int o = 16; o > 0; o >>= 1) ev += __shfl_xor(ev, o, 32);
    float lseA = m + __logf(ev);

    // this lane's direction neighbor (zero logits when out of bounds)
    int ny = y + dx9[d], nx = x + dy9[d];
    bool inB = realRow && ny >= 0 && ny < HH && nx >= 0 && nx < WW;
    float inBf = inB ? 1.0f : 0.0f;
    const float* nb = logits +
        ((size_t)b*HH*WW + (size_t)(inB ? ny : 0)*WW + (inB ? nx : 0)) * CC;

    float bvals[16], ebuf[16];
    v16h A1;
    #pragma unroll
    for (int e = 0; e < 16; ++e) {
      float bv = nb[kclamp[e]] * inBf;    // unconditional load, masked by mul
      float ex = __expf(bv) * kmask[e];   // padded rows/channels -> 0
      bvals[e] = bv; ebuf[e] = ex;
      A1[e] = (_Float16)ex;
    }
    // WMMA #1: S[d] = sum_c exp(b_d[c])   (EXEC all ones here)
    v8f acc = {};
    v8f S = __builtin_amdgcn_wmma_f32_16x16x32_f16(false, A1, false, ones,
                                                   (short)0, acc, false, false);
    if (lane == 0) {           // all D columns identical -> lane0 has rows 0-7
      #pragma unroll
      for (int rr = 0; rr < 8; ++rr) shS[wid][rr] = S[rr];
    }
    asm volatile("s_wait_dscnt 0x0" ::: "memory");
    float Sd = shS[wid][row];                     // unconditional LDS read
    Sd = realRow ? fmaxf(Sd, 1e-30f) : 1.0f;
    float invSd = __builtin_amdgcn_rcpf(Sd);      // v_rcp_f32, no div sequence

    v16h A2;
    #pragma unroll
    for (int e = 0; e < 16; ++e) {
      float ak = shA[wid][kmap[e] & 31];          // unconditional LDS read
      float t = ebuf[e] * invSd * (bvals[e] - ak);
      A2[e] = (_Float16)t;                        // ebuf==0 masks padding
    }
    // WMMA #2: dot[d] = sum_c p_d[c]*(b_d[c]-a[c])
    v8f acc2 = {};
    v8f D = __builtin_amdgcn_wmma_f32_16x16x32_f16(false, A2, false, ones,
                                                   (short)0, acc2, false, false);
    if (lane == 0) {
      float kl[8]; float mk = -3.4e38f;
      #pragma unroll
      for (int i = 0; i < 8; ++i) {
        kl[i] = D[i] + lseA - __logf(fmaxf(S[i], 1e-30f));
        mk = fmaxf(mk, kl[i]);
      }
      float se = 0.0f;
      #pragma unroll
      for (int i = 0; i < 8; ++i) se += __expf(kl[i] - mk);
      float lse = mk + __logf(se);
      float ce = 0.0f;
      #pragma unroll
      for (int i = 0; i < 8; ++i) {
        float lbl = ((i == dirGt) ? 0.8f : 0.0f) + 0.025f;  // LB_SMOOTH=0.2
        ce -= lbl * (kl[i] - lse);
      }
      waveLoss += ce * w;
    }
  }
  if (lane == 0) { shLoss[wid] = waveLoss; shCnt[wid] = waveCnt; }
  __syncthreads();
  if (threadIdx.x == 0) {
    float L = 0.0f; int Cn = 0;
    #pragma unroll
    for (int wv = 0; wv < 8; ++wv) { L += shLoss[wv]; Cn += shCnt[wv]; }
    lossPart[blockIdx.x] = L; cntPart[blockIdx.x] = Cn;
  }
}

// ----------------------------------------- deterministic final reduce
__global__ void finalize_kernel(const float* __restrict__ lossPart,
                                const int* __restrict__ cntPart,
                                float* __restrict__ out) {
  __shared__ float sL[256];
  __shared__ int   sC[256];
  int t = threadIdx.x;
  float L = 0.0f; int Cn = 0;
  for (int i = t; i < NPART; i += 256) { L += lossPart[i]; Cn += cntPart[i]; }
  sL[t] = L; sC[t] = Cn;
  __syncthreads();
  for (int o = 128; o > 0; o >>= 1) {
    if (t < o) { sL[t] += sL[t + o]; sC[t] += sC[t + o]; }
    __syncthreads();
  }
  if (t == 0) out[0] = sL[0] / fmaxf((float)sC[0], 1.0f);
}

// --------------------------------------------------------------- entry
extern "C" void kernel_launch(void* const* d_in, const int* in_sizes, int n_in,
                              void* d_out, int out_size, void* d_ws, size_t ws_size,
                              hipStream_t stream) {
  (void)in_sizes; (void)n_in; (void)out_size; (void)ws_size;
  const float* logits = (const float*)d_in[0];
  const int*   target = (const int*)d_in[1];

  char* ws = (char*)d_ws;
  size_t off = 0;
  auto alloc = [&](size_t bytes) -> void* {
    void* p = ws + off;
    off = (off + bytes + 255) & ~(size_t)255;
    return p;
  };
  float* klmap = (float*)alloc((size_t)PIX * 4);
  float* G     = (float*)alloc((size_t)PIX * 4);
  float* dmap  = (float*)alloc((size_t)PIX * 4);
  unsigned char* gtb  = (unsigned char*)alloc(PIX);
  unsigned char* pred = (unsigned char*)alloc(PIX);
  int*   hist = (int*)alloc(NBINS * 4);
  int*   anyF = (int*)alloc(BB * 4);
  float* eps  = (float*)alloc(4);
  float* lossPart = (float*)alloc(NPART * 4);
  int*   cntPart  = (int*)alloc(NPART * 4);

  init_kernel<<<1, 256, 0, stream>>>(hist, anyF);
  gt_boundary_kernel<<<(PIX + 255) / 256, 256, 0, stream>>>(target, gtb, anyF);
  edt_vertical_kernel<<<(BB * WW + 255) / 256, 256, 0, stream>>>(gtb, G);
  edt_row_kernel<<<BB * HH, WW, 0, stream>>>(G, anyF, dmap);
  kl_map_kernel<<<(PIX + 255) / 256, 256, 0, stream>>>(logits, klmap, hist);
  eps_kernel<<<1, 32, 0, stream>>>(hist, eps);
  threshold_dilate_kernel<<<(PIX + 255) / 256, 256, 0, stream>>>(klmap, eps, pred);
  dir_loss_kernel<<<NPART, 256, 0, stream>>>(logits, dmap, pred, lossPart, cntPart);
  finalize_kernel<<<1, 256, 0, stream>>>(lossPart, cntPart, (float*)d_out);
}
